// CausalSelfAttention_20796231647826
// MI455X (gfx1250) — matrix operand order repaired and checked by reading, not compiled, not run on hardware
//
#include <hip/hip_runtime.h>

typedef __bf16 bf16_t;
typedef __attribute__((ext_vector_type(8)))  __bf16 v8bf;
typedef __attribute__((ext_vector_type(16))) __bf16 v16bf;
typedef __attribute__((ext_vector_type(8)))  float  v8f;
typedef __attribute__((ext_vector_type(4)))  float  v4f;
typedef __attribute__((ext_vector_type(4)))  int    v4i;

#define L2E 1.4426950408889634f

// ---------- CDNA5 async global->LDS copy (guarded; falls back to load+ds_store) ----
#if defined(__AMDGCN__) && __has_builtin(__builtin_amdgcn_global_load_async_to_lds_b128) && \
    __has_builtin(__builtin_amdgcn_s_wait_asynccnt)
#define HAVE_ASYNC 1
#else
#define HAVE_ASYNC 0
#endif

static __device__ __forceinline__ void async_copy16(const bf16_t* g, bf16_t* l) {
#if HAVE_ASYNC
  __builtin_amdgcn_global_load_async_to_lds_b128(
      (__attribute__((address_space(1))) v4i*)(bf16_t*)g,
      (__attribute__((address_space(3))) v4i*)l, 0, 0);
#else
  *(v8bf*)l = *(const v8bf*)g;
#endif
}
#if HAVE_ASYNC
#define WAIT_ASYNC(n) __builtin_amdgcn_s_wait_asynccnt(n)
#else
#define WAIT_ASYNC(n)
#endif

// ---------- WMMA helpers (CDNA5 bf16, D=f32 16x16, K=32) ----------
static __device__ __forceinline__ v8f wmma_bf16(v16bf a, v16bf b, v8f c) {
  return __builtin_amdgcn_wmma_f32_16x16x32_bf16(false, a, false, b, (short)0, c,
                                                 false, false);
}

// A-matrix 16x32 bf16 fragment from row-major src (stride in elements).
static __device__ __forceinline__ v16bf load_a_frag(const bf16_t* base, int stride) {
  const int lane = threadIdx.x & 31;
  const int row  = lane & 15;
  const int kb   = (lane >> 4) << 3;           // 0 or 8
  const bf16_t* p = base + (size_t)row * stride + kb;
  union { v16bf v; v8bf h[2]; } u;
  u.h[0] = *(const v8bf*)(p);                  // K = kb .. kb+7
  u.h[1] = *(const v8bf*)(p + 16);             // K = kb+16 .. kb+23
  return u.v;
}

// B-matrix 32x16 bf16 fragment from column-major B (i.e. B^T row-major).
static __device__ __forceinline__ v16bf load_b_frag(const bf16_t* baseT, int stride) {
  const int lane = threadIdx.x & 31;
  const int col  = lane & 15;
  const int kh   = (lane >> 4) << 3;           // 0 or 8
  const bf16_t* p = baseT + (size_t)col * stride + kh;
  union { v16bf v; v8bf h[2]; } u;
  u.h[0] = *(const v8bf*)(p);
  u.h[1] = *(const v8bf*)(p + 16);
  return u.v;
}

// xor-swizzle reductions within 16-lane row groups (wave32)
#define SWZ(v, imm) __int_as_float(__builtin_amdgcn_ds_swizzle(__float_as_int(v), imm))
static __device__ __forceinline__ float row16_max(float v) {
  v = fmaxf(v, SWZ(v, 0x041F));
  v = fmaxf(v, SWZ(v, 0x081F));
  v = fmaxf(v, SWZ(v, 0x101F));
  v = fmaxf(v, SWZ(v, 0x201F));
  return v;
}
static __device__ __forceinline__ float row16_sum(float v) {
  v += SWZ(v, 0x041F);
  v += SWZ(v, 0x081F);
  v += SWZ(v, 0x101F);
  v += SWZ(v, 0x201F);
  return v;
}

// ---------- problem constants ----------
static constexpr int BATCH = 4;
static constexpr int TSEQ  = 2048;
static constexpr int CDIM  = 1024;
static constexpr int NHEAD = 16;
static constexpr int DHEAD = 64;
static constexpr int MROWS = BATCH * TSEQ;   // 8192
static constexpr int LS    = 88;             // padded LDS row stride (176 B: 16B-aligned,
                                             // 44*r mod 64 banks all-distinct for r=0..15)

// ---------- elementwise f32 -> bf16 ----------
__global__ __launch_bounds__(256) void convert_f32_bf16(const float* __restrict__ in,
                                                        bf16_t* __restrict__ out, int n) {
  int i = (blockIdx.x * 256 + threadIdx.x) * 8;
  if (i + 7 >= n) return;
  v4f a = *(const v4f*)(in + i);
  v4f b = *(const v4f*)(in + i + 4);
  union { v8bf v; bf16_t e[8]; } u;
#pragma unroll
  for (int j = 0; j < 4; ++j) { u.e[j] = (bf16_t)a[j]; u.e[4 + j] = (bf16_t)b[j]; }
  *(v8bf*)(out + i) = u.v;
}

// ---------- tiled transpose + convert: in [K][N] f32 -> out [N][K] bf16 ----------
__global__ __launch_bounds__(256) void transpose_convert(const float* __restrict__ in,
                                                         bf16_t* __restrict__ out,
                                                         int K, int N) {
  __shared__ float tile[32][33];
  const int n0 = blockIdx.x * 32, k0 = blockIdx.y * 32;
  const int tx = threadIdx.x, ty = threadIdx.y;     // block (32,8)
#pragma unroll
  for (int j = 0; j < 4; ++j)
    tile[ty + 8 * j][tx] = in[(size_t)(k0 + ty + 8 * j) * N + n0 + tx];
  __syncthreads();
#pragma unroll
  for (int j = 0; j < 4; ++j)
    out[(size_t)(n0 + ty + 8 * j) * K + k0 + tx] = (bf16_t)tile[tx][ty + 8 * j];
}

// ---------- shared LDS-staged GEMM mainloop ----------
// Block: 256 threads / 8 waves, block tile 128(M) x 128(N), K staged in chunks of 64.
// Waves: wm = wave&3 (4 in M), wn = wave>>2 (2 in N); wave tile 32x64, acc[8] (2Mx4N).
static __device__ __forceinline__ void gemm_mainloop(const bf16_t* __restrict__ Ag, // + m0blk*K
                                                     const bf16_t* __restrict__ Bg, // + n0blk*K
                                                     bf16_t* sA, bf16_t* sB,        // [2][128*LS]
                                                     int K, v8f acc[8], int wm, int wn) {
  const int tid = threadIdx.x;
  const int nstage = K / 64;

  auto issue = [&](int kt, int buf) {
    const size_t gk = (size_t)kt * 64;
    bf16_t* la = sA + buf * (128 * LS);
    bf16_t* lb = sB + buf * (128 * LS);
#pragma unroll
    for (int q = 0; q < 4; ++q) {
      const int cid = q * 256 + tid;           // 1024 16-byte chunks per tile
      const int rr = cid >> 3, cc = (cid & 7) * 8;
      async_copy16(Ag + (size_t)rr * K + gk + cc, la + rr * LS + cc);
      async_copy16(Bg + (size_t)rr * K + gk + cc, lb + rr * LS + cc);
    }
  };

  issue(0, 0);
  for (int kt = 0; kt < nstage; ++kt) {
    const int buf = kt & 1;
    if (kt + 1 < nstage) { issue(kt + 1, buf ^ 1); WAIT_ASYNC(8); }
    else                 { WAIT_ASYNC(0); }
    __syncthreads();                           // stage kt visible to all waves
    const bf16_t* la = sA + buf * (128 * LS) + (wm * 32) * LS;
    const bf16_t* lb = sB + buf * (128 * LS) + (wn * 64) * LS;
#pragma unroll
    for (int kc = 0; kc < 64; kc += 32) {
      v16bf a0 = load_a_frag(la + kc, LS);
      v16bf a1 = load_a_frag(la + 16 * LS + kc, LS);
#pragma unroll
      for (int j = 0; j < 4; ++j) {
        v16bf b = load_b_frag(lb + (16 * j) * LS + kc, LS);
        acc[j]     = wmma_bf16(a0, b, acc[j]);
        acc[4 + j] = wmma_bf16(a1, b, acc[4 + j]);
      }
    }
    __syncthreads();                           // all reads done before buf is re-filled
  }
}

// ---------- QKV GEMM + scatter into q / k / vT ----------
// grid (3072/128=24, 8192/128=64)
__global__ __launch_bounds__(256) void gemm_qkv(const bf16_t* __restrict__ X,
                                                const bf16_t* __restrict__ Wt,   // [3072][1024]
                                                const float*  __restrict__ bias,
                                                bf16_t* __restrict__ qb,   // [B,H,T,Dh]
                                                bf16_t* __restrict__ kb,   // [B,H,T,Dh]
                                                bf16_t* __restrict__ vT) { // [B,H,Dh,T]
  __shared__ bf16_t sA[2 * 128 * LS];
  __shared__ bf16_t sB[2 * 128 * LS];
  const int wave = threadIdx.x >> 5, lane = threadIdx.x & 31;
  const int wm = wave & 3, wn = wave >> 2;
  const int m0blk = blockIdx.y * 128, n0blk = blockIdx.x * 128;

  v8f acc[8] = {};
  gemm_mainloop(X + (size_t)m0blk * CDIM, Wt + (size_t)n0blk * CDIM, sA, sB, CDIM,
                acc, wm, wn);

  const int colf = lane & 15;
  const int rowBase = (lane >> 4) << 3;
#pragma unroll
  for (int mi = 0; mi < 2; ++mi) {
    const int m0 = m0blk + wm * 32 + mi * 16;
    const int bidx = m0 / TSEQ, t0 = m0 % TSEQ;
#pragma unroll
    for (int j = 0; j < 4; ++j) {
      const v8f a = acc[mi * 4 + j];
      const int n = n0blk + wn * 64 + 16 * j + colf;
      const float bv = bias[n];
      const int seg = n >> 10, nn = n & 1023;
      const int h = nn >> 6, d = nn & 63;
      const size_t head = (size_t)(bidx * NHEAD + h);
      if (seg == 0) {            // Q: fold 1/sqrt(Dh)=0.125 (exact in bf16)
#pragma unroll
        for (int i = 0; i < 8; ++i) {
          int t = t0 + rowBase + i;
          qb[(head * TSEQ + t) * DHEAD + d] = (bf16_t)((a[i] + bv) * 0.125f);
        }
      } else if (seg == 1) {     // K row-major
#pragma unroll
        for (int i = 0; i < 8; ++i) {
          int t = t0 + rowBase + i;
          kb[(head * TSEQ + t) * DHEAD + d] = (bf16_t)(a[i] + bv);
        }
      } else {                   // V transposed: contiguous-in-t packed store
        union { v8bf v; bf16_t e[8]; } u;
#pragma unroll
        for (int i = 0; i < 8; ++i) u.e[i] = (bf16_t)(a[i] + bv);
        *(v8bf*)(vT + (head * DHEAD + d) * TSEQ + t0 + rowBase) = u.v;
      }
    }
  }
}

// ---------- Flash attention: grid (B*H=64, T/128=16), block 256 (8 waves) ----------
// K/V 64-key tiles are staged in double-buffered LDS via async copies and shared by
// all 8 waves. Key-tile loop bound is block-uniform (barriers never diverge); waves
// skip compute (whole-wave branch, EXEC stays all-1s) past their causal range.
__global__ __launch_bounds__(256) void attn_kernel(const bf16_t* __restrict__ qb,
                                                   const bf16_t* __restrict__ kb,
                                                   const bf16_t* __restrict__ vT,
                                                   bf16_t* __restrict__ y) { // [B,T,C]
  __shared__ bf16_t sK[2 * 64 * LS];               // [key][d]   tiles
  __shared__ bf16_t sV[2 * 64 * LS];               // [d][key]   tiles
  __shared__ bf16_t pstage[8][16 * 64];            // per-wave P tile (2 KB each)
  const int tid = threadIdx.x;
  const int wave = tid >> 5, lane = tid & 31;
  const int bh = blockIdx.x;
  const int q0blk = blockIdx.y * 128;
  const int q0 = q0blk + wave * 16;
  const bf16_t* Q  = qb + (size_t)bh * TSEQ * DHEAD;
  const bf16_t* Kp = kb + (size_t)bh * TSEQ * DHEAD;
  const bf16_t* Vt = vT + (size_t)bh * DHEAD * TSEQ;

  const v16bf qa0 = load_a_frag(Q + (size_t)q0 * DHEAD + 0,  DHEAD);
  const v16bf qa1 = load_a_frag(Q + (size_t)q0 * DHEAD + 32, DHEAD);

  const int colf = lane & 15;
  const int rowBase = (lane >> 4) << 3;
  v8f   o[4] = {};
  float mr[8], lr[8];
#pragma unroll
  for (int i = 0; i < 8; ++i) { mr[i] = -1e30f; lr[i] = 0.f; }

  bf16_t* pt = pstage[wave];
  const int nkt = (q0blk + 191) >> 6;              // tiles covering keys <= q0blk+127

  // stage issue: K tile 64x64 (512 16B chunks) + V tile 64x64 -> 4 asyncs / thread
  auto issue = [&](int kt, int buf) {
    const int kbase = kt * 64;
    bf16_t* lk = sK + buf * (64 * LS);
    bf16_t* lv = sV + buf * (64 * LS);
#pragma unroll
    for (int q = 0; q < 2; ++q) {
      const int cid = q * 256 + tid;
      const int rr = cid >> 3, cc = (cid & 7) * 8;
      async_copy16(Kp + (size_t)(kbase + rr) * DHEAD + cc, lk + rr * LS + cc);
      async_copy16(Vt + (size_t)rr * TSEQ + kbase + cc,    lv + rr * LS + cc);
    }
  };

  issue(0, 0);
  for (int kt = 0; kt < nkt; ++kt) {
    const int buf = kt & 1;
    const int kbase = kt * 64;
    if (kt + 1 < nkt) { issue(kt + 1, buf ^ 1); WAIT_ASYNC(4); }
    else              { WAIT_ASYNC(0); }
    __syncthreads();
    if (q0 + 15 >= kbase) {                      // wave-uniform causal skip
      const bf16_t* lk = sK + buf * (64 * LS);
      const bf16_t* lv = sV + buf * (64 * LS);
      // ---- S = Q @ K^T (16 x 64) ----
      v8f s[4] = {};
#pragma unroll
      for (int j = 0; j < 4; ++j) {
        const bf16_t* kp = lk + (16 * j) * LS;
        s[j] = wmma_bf16(qa0, load_b_frag(kp + 0,  LS), s[j]);
        s[j] = wmma_bf16(qa1, load_b_frag(kp + 32, LS), s[j]);
      }
      if (kbase + 63 > q0) {
#pragma unroll
        for (int j = 0; j < 4; ++j)
#pragma unroll
          for (int i = 0; i < 8; ++i)
            if (kbase + 16 * j + colf > q0 + rowBase + i) s[j][i] = -1e30f;
      }
      // ---- online softmax ----
      float mnew[8], rowsum[8];
#pragma unroll
      for (int i = 0; i < 8; ++i) {
        float v = fmaxf(fmaxf(s[0][i], s[1][i]), fmaxf(s[2][i], s[3][i]));
        mnew[i] = fmaxf(mr[i], row16_max(v));
      }
#pragma unroll
      for (int i = 0; i < 8; ++i) {
        float ps = 0.f;
#pragma unroll
        for (int j = 0; j < 4; ++j) {
          float p = __builtin_amdgcn_exp2f((s[j][i] - mnew[i]) * L2E);
          s[j][i] = p;
          ps += p;
        }
        rowsum[i] = row16_sum(ps);
      }
#pragma unroll
      for (int i = 0; i < 8; ++i) {
        float corr = __builtin_amdgcn_exp2f((mr[i] - mnew[i]) * L2E);
        lr[i] = lr[i] * corr + rowsum[i];
        mr[i] = mnew[i];
#pragma unroll
        for (int f = 0; f < 4; ++f) o[f][i] *= corr;
      }
      // ---- stage P through wave-private LDS: D-layout -> A-layout ----
      __builtin_amdgcn_wave_barrier();
#pragma unroll
      for (int j = 0; j < 4; ++j)
#pragma unroll
        for (int i = 0; i < 8; ++i)
          pt[(rowBase + i) * 64 + 16 * j + colf] = (bf16_t)s[j][i];
      __builtin_amdgcn_wave_barrier();
      const v16bf pa0 = load_a_frag(pt, 64);
      const v16bf pa1 = load_a_frag(pt + 32, 64);
      // ---- O += P @ V ----
#pragma unroll
      for (int f = 0; f < 4; ++f) {
        const bf16_t* vp = lv + (16 * f) * LS;
        o[f] = wmma_bf16(pa0, load_b_frag(vp + 0,  LS), o[f]);
        o[f] = wmma_bf16(pa1, load_b_frag(vp + 32, LS), o[f]);
      }
    }
    __syncthreads();                             // all reads done before buf refill
  }

  const int b = bh / NHEAD, h = bh % NHEAD;
#pragma unroll
  for (int f = 0; f < 4; ++f)
#pragma unroll
    for (int i = 0; i < 8; ++i) {
      const size_t t = (size_t)q0 + rowBase + i;
      y[((size_t)b * TSEQ + t) * CDIM + h * DHEAD + 16 * f + colf] =
          (bf16_t)(o[f][i] / lr[i]);
    }
}

// ---------- output projection: grid (1024/128=8, 64) ----------
__global__ __launch_bounds__(256) void gemm_proj(const bf16_t* __restrict__ Y,
                                                 const bf16_t* __restrict__ Wt,
                                                 const float*  __restrict__ bias,
                                                 float* __restrict__ out) {
  __shared__ bf16_t sA[2 * 128 * LS];
  __shared__ bf16_t sB[2 * 128 * LS];
  const int wave = threadIdx.x >> 5, lane = threadIdx.x & 31;
  const int wm = wave & 3, wn = wave >> 2;
  const int m0blk = blockIdx.y * 128, n0blk = blockIdx.x * 128;

  v8f acc[8] = {};
  gemm_mainloop(Y + (size_t)m0blk * CDIM, Wt + (size_t)n0blk * CDIM, sA, sB, CDIM,
                acc, wm, wn);

  const int colf = lane & 15;
  const int rowBase = (lane >> 4) << 3;
#pragma unroll
  for (int mi = 0; mi < 2; ++mi) {
    const int m0 = m0blk + wm * 32 + mi * 16;
#pragma unroll
    for (int j = 0; j < 4; ++j) {
      const int n = n0blk + wn * 64 + 16 * j + colf;
      const float bv = bias[n];
#pragma unroll
      for (int i = 0; i < 8; ++i)
        out[(size_t)(m0 + rowBase + i) * CDIM + n] = acc[mi * 4 + j][i] + bv;
    }
  }
}

// ---------- launch ----------
extern "C" void kernel_launch(void* const* d_in, const int* in_sizes, int n_in,
                              void* d_out, int out_size, void* d_ws, size_t ws_size,
                              hipStream_t stream) {
  const float* x      = (const float*)d_in[0];
  const float* W_attn = (const float*)d_in[1];
  const float* b_attn = (const float*)d_in[2];
  const float* W_proj = (const float*)d_in[3];
  const float* b_proj = (const float*)d_in[4];
  float* out = (float*)d_out;

  char* ws = (char*)d_ws;
  size_t off = 0;
  auto alloc = [&](size_t elems) { bf16_t* p = (bf16_t*)(ws + off); off += elems * sizeof(bf16_t); return p; };
  bf16_t* xb  = alloc((size_t)MROWS * CDIM);
  bf16_t* wat = alloc((size_t)3 * CDIM * CDIM);
  bf16_t* wpt = alloc((size_t)CDIM * CDIM);
  bf16_t* qb  = alloc((size_t)MROWS * CDIM);
  bf16_t* kb  = alloc((size_t)MROWS * CDIM);
  bf16_t* vt  = alloc((size_t)MROWS * CDIM);
  bf16_t* yb  = alloc((size_t)MROWS * CDIM);
  (void)ws_size; (void)n_in; (void)in_sizes; (void)out_size;

  const int nx = MROWS * CDIM;
  convert_f32_bf16<<<nx / (256 * 8), 256, 0, stream>>>(x, xb, nx);
  transpose_convert<<<dim3(3 * CDIM / 32, CDIM / 32), dim3(32, 8), 0, stream>>>(W_attn, wat, CDIM, 3 * CDIM);
  transpose_convert<<<dim3(CDIM / 32, CDIM / 32), dim3(32, 8), 0, stream>>>(W_proj, wpt, CDIM, CDIM);

  gemm_qkv<<<dim3(3 * CDIM / 128, MROWS / 128), 256, 0, stream>>>(xb, wat, b_attn, qb, kb, vt);
  attn_kernel<<<dim3(BATCH * NHEAD, TSEQ / 128), 256, 0, stream>>>(qb, kb, vt, yb);
  gemm_proj<<<dim3(CDIM / 128, MROWS / 128), 256, 0, stream>>>(yb, wpt, b_proj, out);
}
